// StrandGeneratorSiren_60172491817620
// MI455X (gfx1250) — matrix operand order repaired and checked by reading, not compiled, not run on hardware
//
#include <hip/hip_runtime.h>
#include <hip/hip_bf16.h>

#define IN_CH    64
#define MOD_H    512
#define SIREN_H  512
#define NUM_PTS  100
#define NV       99
#define N_STRANDS 1024
#define ROWS     (N_STRANDS * NV)
#define ZK       576   // IN_CH + MOD_H

typedef __attribute__((ext_vector_type(16))) __bf16 v16bf;
typedef __attribute__((ext_vector_type(8)))  float  v8f;

// ---------------------------------------------------------------------------
// LDS tile: 16 rows x 512 bf16, XOR-swizzled in 16-byte (8-elem) chunks so
// that the 16 per-row A-fragment loads (one row per lane) hit distinct banks.
// ---------------------------------------------------------------------------
__device__ __forceinline__ int lds_off(int m, int k) {
  return m * 512 + ((((k >> 3) ^ m) & 63) << 3) + (k & 7);
}

__device__ __forceinline__ v8f zero8() {
  v8f z = {0.f, 0.f, 0.f, 0.f, 0.f, 0.f, 0.f, 0.f};
  return z;
}

// B fragment: 32(K) x 16(N) bf16 tile of row-major W[N][K].
// Lane holds column n = lane&15; elems 0..7 -> K=kk+8*half..+7, elems 8..15
// -> K=kk+16+8*half..+7  -> two 16B global loads.
__device__ __forceinline__ v16bf load_b_frag(const __bf16* __restrict__ W,
                                             int Kdim, int n, int kk, int half) {
  const __bf16* p = W + n * Kdim + kk + (half << 3);
  v16bf b;
  ((uint4*)&b)[0] = *(const uint4*)p;
  ((uint4*)&b)[1] = *(const uint4*)(p + 16);
  return b;
}

// A fragment: 16(M) x 32(K). Lane holds row m = lane&15; lanes 0-15:
// elems 0..7 -> K=kk..kk+7, elems 8..15 -> K=kk+16..kk+23; lanes 16-31 +8.
__device__ __forceinline__ v16bf load_a_lds(const __bf16* buf, int m, int kk, int half) {
  v16bf a;
  ((uint4*)&a)[0] = *(const uint4*)(buf + lds_off(m, kk + (half << 3)));
  ((uint4*)&a)[1] = *(const uint4*)(buf + lds_off(m, kk + 16 + (half << 3)));
  return a;
}

__device__ __forceinline__ v8f wmma_bf16(v16bf a, v16bf b, v8f c) {
  // (neg_a, A, neg_b, B, c_mod, C, reuse_a, reuse_b)
  return __builtin_amdgcn_wmma_f32_16x16x32_bf16(false, a, false, b, (short)0, c,
                                                 false, false);
}

// Fast silu: x * v_rcp_f32(1 + v_exp(-x)); avoids IEEE div expansion.
__device__ __forceinline__ float fast_silu(float x) {
  return x * __builtin_amdgcn_rcpf(1.0f + __expf(-x));
}

// ---------------------------------------------------------------------------
// One fused layer:  h_mod = silu(z @ Wm^T + bm)
//                   s     = sin([30*] (h_s @ Ws^T + bs))
//                   h_s'  = (1 - h_mod) * s
// z cols 0..63 come from registers (A0/A1 = z_initial), cols 64..575 from
// hmPrev (LDS). Layer 0's siren branch has K=1 and is computed analytically.
// ---------------------------------------------------------------------------
template <int L>
__device__ __forceinline__ void layer(
    int row0, int half, int ln, v16bf A0, v16bf A1,
    const __bf16* __restrict__ MW, const float* __restrict__ MB,
    const __bf16* __restrict__ SW, const float* __restrict__ SB,
    const float* __restrict__ s0w, const float* __restrict__ s0b,
    const __bf16* hmPrev, __bf16* hmNext,
    const __bf16* hsPrev, __bf16* hsNext) {
  const int Kmod = (L == 0) ? IN_CH : ZK;
  for (int nt = 0; nt < MOD_H / 16; ++nt) {
    const int col = nt * 16 + ln;

    v8f accM = zero8();
    accM = wmma_bf16(A0, load_b_frag(MW, Kmod, col, 0, half), accM);
    accM = wmma_bf16(A1, load_b_frag(MW, Kmod, col, 32, half), accM);
    if (L > 0) {
#pragma unroll 4
      for (int kk = 64; kk < ZK; kk += 32)
        accM = wmma_bf16(load_a_lds(hmPrev, ln, kk - 64, half),
                         load_b_frag(MW, ZK, col, kk, half), accM);
    }

    v8f accS = zero8();
    if (L > 0) {
#pragma unroll 4
      for (int kk = 0; kk < SIREN_H; kk += 32)
        accS = wmma_bf16(load_a_lds(hsPrev, ln, kk, half),
                         load_b_frag(SW, SIREN_H, col, kk, half), accS);
    }

    const float bm = MB[col];
    float w0 = 0.f, b0 = 0.f, bs = 0.f;
    if (L == 0) { w0 = s0w[col]; b0 = s0b[col]; } else { bs = SB[col]; }

#pragma unroll
    for (int r = 0; r < 8; ++r) {
      const int m = r + (half << 3);           // output row within tile
      const float hm = fast_silu(accM[r] + bm);
      float s;
      if (L == 0) {
        const float t = (float)((row0 + m) % NV) * (1.0f / 99.0f);
        s = __sinf(30.0f * (t * w0 + b0));     // SIREN first layer, K=1
      } else {
        s = __sinf(accS[r] + bs);
      }
      hmNext[lds_off(m, col)] = (__bf16)hm;
      hsNext[lds_off(m, col)] = (__bf16)((1.0f - hm) * s);
    }
  }
  __syncthreads();  // make cross-lane LDS writes visible before next layer
}

// ---------------------------------------------------------------------------
// Main fused kernel: one wave (32 threads) per block, 16 rows per block.
// ---------------------------------------------------------------------------
__global__ __launch_bounds__(32) void strand_siren_fused(
    const float* __restrict__ sf,
    const __bf16* __restrict__ mw0, const __bf16* __restrict__ mw1,
    const __bf16* __restrict__ mw2,
    const __bf16* __restrict__ sw1, const __bf16* __restrict__ sw2,
    const float* __restrict__ s0w,
    const float* __restrict__ mb0, const float* __restrict__ mb1,
    const float* __restrict__ mb2,
    const float* __restrict__ s0b, const float* __restrict__ sb1,
    const float* __restrict__ sb2,
    const __bf16* __restrict__ decw,   // 16 x 512 bf16, rows 3..15 zero
    const float* __restrict__ decb,
    float* __restrict__ hair) {
  __shared__ __bf16 smem[4 * 16 * 512];  // 64 KB: hmA,hmB,hsA,hsB ping-pong
  __bf16* hmA = smem + 0 * 8192;
  __bf16* hmB = smem + 1 * 8192;
  __bf16* hsA = smem + 2 * 8192;
  __bf16* hsB = smem + 3 * 8192;

  const int lane = threadIdx.x & 31;
  const int half = lane >> 4;
  const int ln   = lane & 15;
  const int row0 = blockIdx.x * 16;

  // z_initial: per-lane A fragments (row m = ln), constant across layers.
  const float* zr = sf + ((row0 + ln) / NV) * IN_CH;
  v16bf A0, A1;
  const int kb = half << 3;
#pragma unroll
  for (int j = 0; j < 8; ++j) {
    A0[j]     = (__bf16)zr[kb + j];
    A0[j + 8] = (__bf16)zr[kb + 16 + j];
    A1[j]     = (__bf16)zr[kb + 32 + j];
    A1[j + 8] = (__bf16)zr[kb + 48 + j];
  }

  layer<0>(row0, half, ln, A0, A1, mw0, mb0, (const __bf16*)nullptr,
           (const float*)nullptr, s0w, s0b, nullptr, hmB, nullptr, hsA);
  layer<1>(row0, half, ln, A0, A1, mw1, mb1, sw1, sb1, s0w, s0b,
           hmB, hmA, hsA, hsB);
  layer<2>(row0, half, ln, A0, A1, mw2, mb2, sw2, sb2, s0w, s0b,
           hmA, hmB, hsB, hsA);

  // Decoder as one 16x16 WMMA tile (cols 3..15 of B are zero):
  // hair_dir = (h_siren @ Wdec^T + bdec) * 0.01, h_siren in hsA.
  {
    v8f acc = zero8();
#pragma unroll 4
    for (int kk = 0; kk < SIREN_H; kk += 32)
      acc = wmma_bf16(load_a_lds(hsA, ln, kk, half),
                      load_b_frag(decw, SIREN_H, ln, kk, half), acc);
    if (ln < 3) {
      const float bd = decb[ln];
#pragma unroll
      for (int r = 0; r < 8; ++r) {
        const int m = r + (half << 3);
        hair[(row0 + m) * 3 + ln] = 0.01f * (acc[r] + bd);
      }
    }
  }
}

// ---------------------------------------------------------------------------
// Weight-norm prologues: w[o][k] = v[o][k] * g[o] / ||v[o]||  (one wave/row)
// ---------------------------------------------------------------------------
__global__ __launch_bounds__(32) void wnorm_bf16_k(const float* __restrict__ v,
                                                   const float* __restrict__ g,
                                                   int K, __bf16* __restrict__ out) {
  const int o = blockIdx.x;
  const float* row = v + (size_t)o * K;
  float ss = 0.f;
  for (int k = threadIdx.x; k < K; k += 32) { float x = row[k]; ss += x * x; }
#pragma unroll
  for (int off = 16; off > 0; off >>= 1) ss += __shfl_xor(ss, off);
  const float sc = g[o] * rsqrtf(ss);
  for (int k = threadIdx.x; k < K; k += 32)
    out[(size_t)o * K + k] = (__bf16)(row[k] * sc);
}

__global__ __launch_bounds__(32) void wnorm_f32_k(const float* __restrict__ v,
                                                  const float* __restrict__ g,
                                                  int K, float* __restrict__ out) {
  const int o = blockIdx.x;
  const float* row = v + (size_t)o * K;
  float ss = 0.f;
  for (int k = threadIdx.x; k < K; k += 32) { float x = row[k]; ss += x * x; }
#pragma unroll
  for (int off = 16; off > 0; off >>= 1) ss += __shfl_xor(ss, off);
  const float sc = g[o] * rsqrtf(ss);
  for (int k = threadIdx.x; k < K; k += 32)
    out[(size_t)o * K + k] = row[k] * sc;
}

// Decoder weights: normalize 3 real rows, zero-pad to 16 rows of bf16.
__global__ __launch_bounds__(32) void wnorm_dec_k(const float* __restrict__ v,
                                                  const float* __restrict__ g,
                                                  __bf16* __restrict__ out) {
  const int o = blockIdx.x;  // 0..15
  if (o < 3) {
    const float* row = v + (size_t)o * SIREN_H;
    float ss = 0.f;
    for (int k = threadIdx.x; k < SIREN_H; k += 32) { float x = row[k]; ss += x * x; }
#pragma unroll
    for (int off = 16; off > 0; off >>= 1) ss += __shfl_xor(ss, off);
    const float sc = g[o] * rsqrtf(ss);
    for (int k = threadIdx.x; k < SIREN_H; k += 32)
      out[(size_t)o * SIREN_H + k] = (__bf16)(row[k] * sc);
  } else {
    for (int k = threadIdx.x; k < SIREN_H; k += 32)
      out[(size_t)o * SIREN_H + k] = (__bf16)0.0f;
  }
}

// Per-strand cumsum with zero prefix: out (1024, 100, 3)
__global__ void cumsum_k(const float* __restrict__ hair, float* __restrict__ out) {
  const int id = blockIdx.x * blockDim.x + threadIdx.x;
  if (id >= N_STRANDS * 3) return;
  const int n = id / 3, c = id % 3;
  out[(n * NUM_PTS) * 3 + c] = 0.f;
  float acc = 0.f;
  const float* src = hair + n * NV * 3 + c;
  float* dst = out + (n * NUM_PTS + 1) * 3 + c;
  for (int j = 0; j < NV; ++j) { acc += src[j * 3]; dst[j * 3] = acc; }
}

extern "C" void kernel_launch(void* const* d_in, const int* in_sizes, int n_in,
                              void* d_out, int out_size, void* d_ws, size_t ws_size,
                              hipStream_t stream) {
  (void)in_sizes; (void)n_in; (void)out_size; (void)ws_size;
  const float* sf  = (const float*)d_in[0];
  const float* mv0 = (const float*)d_in[1];
  const float* mg0 = (const float*)d_in[2];
  const float* mb0 = (const float*)d_in[3];
  const float* mv1 = (const float*)d_in[4];
  const float* mg1 = (const float*)d_in[5];
  const float* mb1 = (const float*)d_in[6];
  const float* mv2 = (const float*)d_in[7];
  const float* mg2 = (const float*)d_in[8];
  const float* mb2 = (const float*)d_in[9];
  const float* sv0 = (const float*)d_in[10];
  const float* sg0 = (const float*)d_in[11];
  const float* sb0 = (const float*)d_in[12];
  const float* sv1 = (const float*)d_in[13];
  const float* sg1 = (const float*)d_in[14];
  const float* sb1 = (const float*)d_in[15];
  const float* sv2 = (const float*)d_in[16];
  const float* sg2 = (const float*)d_in[17];
  const float* sb2 = (const float*)d_in[18];
  const float* dv  = (const float*)d_in[19];
  const float* dg  = (const float*)d_in[20];
  const float* db  = (const float*)d_in[21];

  // Workspace layout (~3.5 MB)
  uint8_t* ws = (uint8_t*)d_ws;
  size_t off = 0;
  auto alloc = [&](size_t bytes) -> void* {
    void* p = ws + off;
    off = (off + bytes + 255) & ~(size_t)255;
    return p;
  };
  __bf16* mw0  = (__bf16*)alloc((size_t)MOD_H * IN_CH * 2);
  __bf16* mw1  = (__bf16*)alloc((size_t)MOD_H * ZK * 2);
  __bf16* mw2  = (__bf16*)alloc((size_t)MOD_H * ZK * 2);
  __bf16* sw1b = (__bf16*)alloc((size_t)SIREN_H * SIREN_H * 2);
  __bf16* sw2b = (__bf16*)alloc((size_t)SIREN_H * SIREN_H * 2);
  float*  s0w  = (float*)alloc((size_t)SIREN_H * 4);
  __bf16* decw = (__bf16*)alloc((size_t)16 * SIREN_H * 2);
  float*  hair = (float*)alloc((size_t)ROWS * 3 * 4);

  // Weight-norm prologues (weights are tiny; dominated by main kernel).
  wnorm_bf16_k<<<MOD_H, 32, 0, stream>>>(mv0, mg0, IN_CH, mw0);
  wnorm_bf16_k<<<MOD_H, 32, 0, stream>>>(mv1, mg1, ZK, mw1);
  wnorm_bf16_k<<<MOD_H, 32, 0, stream>>>(mv2, mg2, ZK, mw2);
  wnorm_bf16_k<<<SIREN_H, 32, 0, stream>>>(sv1, sg1, SIREN_H, sw1b);
  wnorm_bf16_k<<<SIREN_H, 32, 0, stream>>>(sv2, sg2, SIREN_H, sw2b);
  wnorm_f32_k<<<SIREN_H, 32, 0, stream>>>(sv0, sg0, 1, s0w);
  wnorm_dec_k<<<16, 32, 0, stream>>>(dv, dg, decw);

  // Fused MLP over all 101,376 rows: 16 rows per (single-wave) block.
  strand_siren_fused<<<ROWS / 16, 32, 0, stream>>>(
      sf, mw0, mw1, mw2, sw1b, sw2b, s0w, mb0, mb1, mb2, sb0, sb1, sb2,
      decw, db, hair);

  // Per-strand prefix sum + zero start point -> d_out (1024,100,3) f32.
  cumsum_k<<<(N_STRANDS * 3 + 127) / 128, 128, 0, stream>>>(hair, (float*)d_out);
}
